// GatedDeltaNetMixer_31215822307435
// MI455X (gfx1250) — compile-verified, matrix operand order checked by reading, not run on hardware
//
#include <hip/hip_runtime.h>
#include <stdint.h>

// ---------------- problem constants ----------------
#define BB 2
#define TT 2048
#define DD 1024
#define HH 8
#define KD 128
#define VD 256
#define KEY_DIM 1024   // H*KD
#define VAL_DIM 2048   // H*VD
#define NCAT 6144      // q(1024) | k(1024) | v(2048) | g(2048)
#define ROWS (BB*TT)   // 4096
#define KS 4

typedef __attribute__((ext_vector_type(16))) __bf16 v16bf;
typedef __attribute__((ext_vector_type(8)))  float  v8f;
typedef __attribute__((ext_vector_type(8)))  unsigned int v8u;

// fp32 -> bf16 (round to nearest even)
__device__ __forceinline__ unsigned short f2bf(float f) {
  unsigned u = __builtin_bit_cast(unsigned, f);
  u += 0x7FFFu + ((u >> 16) & 1u);
  return (unsigned short)(u >> 16);
}
__device__ __forceinline__ unsigned pack2bf(float lo, float hi) {
  return (unsigned)f2bf(lo) | ((unsigned)f2bf(hi) << 16);
}
__device__ __forceinline__ float silu(float x) { return x / (1.f + __expf(-x)); }

// ---------------- fp32 -> bf16 row-major conversion (activations) ----------------
__global__ void cvt_bf16(const float* __restrict__ src, __bf16* __restrict__ dst, int n) {
  int i = blockIdx.x * blockDim.x + threadIdx.x;
  if (i < n) dst[i] = __builtin_bit_cast(__bf16, f2bf(src[i]));
}

// ---------------- weight pre-pack: fp32 [rows x cols] -> WMMA-native K-pair panels ----------
// dst layout: for each 32-row K-tile kt, a panel [N][16] of uints, each uint = bf16 pair
// (W[kt*32+2*kp][c], W[kt*32+2*kp+1][c]).  dst index = ((kp_global>>4)*N + colOff + c)*16 + (kp_global&15)
__global__ void pack_w(const float* __restrict__ W, unsigned* __restrict__ dst,
                       int rows, int cols, int colOff, int N) {
  int id = blockIdx.x * blockDim.x + threadIdx.x;
  if (id >= (rows / 2) * cols) return;
  int kpg = id / cols, c = id % cols;
  float lo = W[(size_t)(2 * kpg) * cols + c];
  float hi = W[(size_t)(2 * kpg + 1) * cols + c];
  dst[((size_t)(kpg >> 4) * N + colOff + c) * 16 + (kpg & 15)] = pack2bf(lo, hi);
}

// ---------------- bf16 WMMA GEMM: C[MxN] = A_bf16[MxK] @ Bpacked ----------------
// block = 256 threads (8 waves). Block tile 128x128, BK=32, double-buffered LDS,
// both tiles streamed with global_load_async_to_lds_b128 (fallback: plain copies).
// Wave grid 4x2; each wave: 2 (M) x 4 (N) tiles of v_wmma_f32_16x16x32_bf16.

#if __has_builtin(__builtin_amdgcn_global_load_async_to_lds_b128) && \
    __has_builtin(__builtin_amdgcn_s_wait_asynccnt)
#define HAS_ASYNC_LDS 1
typedef int v4i_t __attribute__((ext_vector_type(4)));
typedef __attribute__((address_space(1))) v4i_t g_v4i;   // global int4
typedef __attribute__((address_space(3))) v4i_t l_v4i;   // LDS int4
#else
#define HAS_ASYNC_LDS 0
#endif

__global__ void __launch_bounds__(256)
gemm_bf16_wmma(const __bf16* __restrict__ A, const unsigned* __restrict__ Bp,
               float* __restrict__ C, int M, int N, int K) {
  __shared__ __align__(16) unsigned AsmB[2][128 * 16];  // [row][kpair]  (8KB each)
  __shared__ __align__(16) unsigned BsmB[2][128 * 16];  // [col][kpair]  (8KB each)

  const int tid  = threadIdx.x;
  const int lane = tid & 31;
  const int wave = tid >> 5;
  const int wm = wave >> 1;        // 0..3
  const int wn = wave & 1;         // 0..1
  const int m  = lane & 15;
  const int half = lane >> 4;      // 0/1

  const int m0 = blockIdx.y * 128;
  const int n0 = blockIdx.x * 128;
  const unsigned* Au = (const unsigned*)A;   // packed bf16 pairs, row stride K/2
  const int nk = K >> 5;

  // per-lane K-pair indices for the 16x32 bf16 A/B layout (ISA 7.12.2)
  int kp[8];
#pragma unroll
  for (int v = 0; v < 8; ++v) {
    int base = (v < 4) ? v : (8 + (v - 4));  // kidx/2
    kp[v] = base + 4 * half;
  }

  // stage one 128x32 A tile + one 128x32 B tile into LDS buffer `sel`
  auto issue = [&](int kt, int sel) {
#pragma unroll
    for (int p = 0; p < 2; ++p) {
      int u = tid + p * 256;                 // 512 x 16B units per tile
      int row = u >> 2, q = u & 3;
      const unsigned* as = Au + (size_t)(m0 + row) * (K >> 1) + kt * 16 + q * 4;
      unsigned* ad = &AsmB[sel][row * 16 + q * 4];
      const unsigned* bs = Bp + ((size_t)kt * N + n0) * 16 + u * 4;
      unsigned* bd = &BsmB[sel][u * 4];
#if HAS_ASYNC_LDS
      __builtin_amdgcn_global_load_async_to_lds_b128((g_v4i*)as, (l_v4i*)ad, 0, 0);
      __builtin_amdgcn_global_load_async_to_lds_b128((g_v4i*)bs, (l_v4i*)bd, 0, 0);
#else
      *(uint4*)ad = *(const uint4*)as;
      *(uint4*)bd = *(const uint4*)bs;
#endif
    }
  };

  v8f acc[2][4] = {};

  issue(0, 0);
  if (nk > 1) issue(1, 1);

  for (int k = 0; k < nk; ++k) {
    const int sel = k & 1;
#if HAS_ASYNC_LDS
    if (k + 1 < nk) __builtin_amdgcn_s_wait_asynccnt(4);   // stage k complete, k+1 in flight
    else            __builtin_amdgcn_s_wait_asynccnt(0);
#endif
    __syncthreads();

    v16bf afr[2];
#pragma unroll
    for (int mt = 0; mt < 2; ++mt) {
      int row = wm * 32 + mt * 16 + m;
      v8u au;
#pragma unroll
      for (int v = 0; v < 8; ++v) au[v] = AsmB[sel][row * 16 + kp[v]];
      afr[mt] = __builtin_bit_cast(v16bf, au);
    }
    v16bf bfr[4];
#pragma unroll
    for (int nt = 0; nt < 4; ++nt) {
      int col = wn * 64 + nt * 16 + m;
      v8u bu;
#pragma unroll
      for (int v = 0; v < 8; ++v) bu[v] = BsmB[sel][col * 16 + kp[v]];
      bfr[nt] = __builtin_bit_cast(v16bf, bu);
    }
#pragma unroll
    for (int mt = 0; mt < 2; ++mt)
#pragma unroll
      for (int nt = 0; nt < 4; ++nt)
        acc[mt][nt] = __builtin_amdgcn_wmma_f32_16x16x32_bf16(
            false, afr[mt], false, bfr[nt], (short)0, acc[mt][nt], false, false);

    __syncthreads();                        // reads of buf[sel] done
    if (k + 2 < nk) issue(k + 2, sel);      // refill the buffer just freed
  }

  // ---- store C (16x16 f32 layout: vgpr r -> M = r + 8*half, N = lane&15) ----
#pragma unroll
  for (int mt = 0; mt < 2; ++mt)
#pragma unroll
    for (int nt = 0; nt < 4; ++nt) {
      int gn = n0 + wn * 64 + nt * 16 + m;
#pragma unroll
      for (int r = 0; r < 8; ++r) {
        int gm = m0 + wm * 32 + mt * 16 + half * 8 + r;
        C[(size_t)gm * N + gn] = acc[mt][nt][r];
      }
    }
}

// ---------------- alpha / beta: skinny projections + gate math ----------------
__global__ void ab_kernel(const float* __restrict__ x, const float* __restrict__ Wa,
                          const float* __restrict__ Wb, const float* __restrict__ A_log,
                          const float* __restrict__ dt_bias,
                          float* __restrict__ alpha, float* __restrict__ beta) {
  int id = blockIdx.x * blockDim.x + threadIdx.x;
  if (id >= ROWS * HH) return;
  int row = id / HH, h = id % HH;
  const float* xr = x + (size_t)row * DD;
  float sa = 0.f, sb = 0.f;
  for (int d = 0; d < DD; ++d) {
    float xv = xr[d];
    sa += xv * Wa[d * HH + h];
    sb += xv * Wb[d * HH + h];
  }
  float z = sa + dt_bias[h];
  float sp = (z > 20.f) ? z : log1pf(__expf(z));     // softplus
  alpha[id] = __expf(-__expf(A_log[h]) * sp);
  beta[id]  = 2.f / (1.f + __expf(-sb));
}

// ---------------- causal dw-conv + SiLU + per-head L2 norm (q / k) ----------------
__global__ void conv_qk_kernel(const float* __restrict__ Y1, int colOff,
                               const float* __restrict__ convW, float* __restrict__ out) {
  __shared__ float red[128];
  int blk = blockIdx.x;            // bt * H + h
  int h = blk % HH, bt = blk / HH;
  int t = bt % TT;
  int c = h * KD + threadIdx.x;    // channel within KEY_DIM
  float s = 0.f;
#pragma unroll
  for (int i = 0; i < KS; ++i) {
    int dt = t - (KS - 1) + i;
    if (dt >= 0) s += convW[c * KS + i] * Y1[(size_t)(bt - (KS - 1) + i) * NCAT + colOff + c];
  }
  s = silu(s);
  red[threadIdx.x] = s * s;
  for (int st = 64; st > 0; st >>= 1) {
    __syncthreads();
    if (threadIdx.x < st) red[threadIdx.x] += red[threadIdx.x + st];
  }
  __syncthreads();
  float nrm = sqrtf(red[0]);
  out[(size_t)bt * KEY_DIM + c] = s / fmaxf(nrm, 1e-6f);
}

// ---------------- causal dw-conv + SiLU (v) ----------------
__global__ void conv_v_kernel(const float* __restrict__ Y1, const float* __restrict__ convW,
                              float* __restrict__ out) {
  int idx = blockIdx.x * blockDim.x + threadIdx.x;
  if (idx >= ROWS * VAL_DIM) return;
  int bt = idx / VAL_DIM, c = idx % VAL_DIM;
  int t = bt % TT;
  float s = 0.f;
#pragma unroll
  for (int i = 0; i < KS; ++i) {
    int dt = t - (KS - 1) + i;
    if (dt >= 0) s += convW[c * KS + i] * Y1[(size_t)(bt - (KS - 1) + i) * NCAT + 2048 + c];
  }
  out[idx] = silu(s);
}

// ---------------- delta-rule scan: one block per (b,h), state register-resident ----------------
// 1024 threads: thread = (v_row = tid>>2, k_quarter = tid&3), 32 state floats each.
__global__ void __launch_bounds__(1024)
scan_kernel(const float* __restrict__ qc, const float* __restrict__ kc,
            const float* __restrict__ vc, const float* __restrict__ alpha,
            const float* __restrict__ beta, float* __restrict__ O) {
  __shared__ float ksh[KD], qsh[KD], vsh[VD], sAB[2];
  const int tid = threadIdx.x;
  const int b = blockIdx.x / HH, h = blockIdx.x % HH;
  const int vrow = tid >> 2, kq = tid & 3, kbase = kq * 32;

  float s[32];
#pragma unroll
  for (int j = 0; j < 32; ++j) s[j] = 0.f;

  for (int t = 0; t < TT; ++t) {
    int bt = b * TT + t;
    if (tid < KD)                      ksh[tid]       = kc[(size_t)bt * KEY_DIM + h * KD + tid];
    else if (tid < 2 * KD)             qsh[tid - KD]  = qc[(size_t)bt * KEY_DIM + h * KD + (tid - KD)];
    else if (tid < 2 * KD + VD)        vsh[tid - 256] = vc[(size_t)bt * VAL_DIM + h * VD + (tid - 256)];
    else if (tid == 512)               sAB[0] = alpha[bt * HH + h];
    else if (tid == 513)               sAB[1] = beta[bt * HH + h];
    __syncthreads();

    // retained = state_prev @ k
    float r = 0.f;
#pragma unroll
    for (int j = 0; j < 32; ++j) r += s[j] * ksh[kbase + j];
    r += __shfl_xor(r, 1, 32);
    r += __shfl_xor(r, 2, 32);

    float aa = sAB[0], bb = sAB[1], vv = vsh[vrow];
    float coef = bb * vv - aa * bb * r;      // state = a*state + coef*k

    float o = 0.f;
#pragma unroll
    for (int j = 0; j < 32; ++j) {
      float kj = ksh[kbase + j];
      s[j] = aa * s[j] + coef * kj;
      o += s[j] * qsh[kbase + j];
    }
    o += __shfl_xor(o, 1, 32);
    o += __shfl_xor(o, 2, 32);
    if (kq == 0) O[(size_t)bt * VAL_DIM + h * VD + vrow] = o;
    __syncthreads();   // before LDS rewrite next step
  }
}

// ---------------- RMS norm over VD + SiLU gate -> bf16 (A of final GEMM) ----------------
__global__ void gate_rms_kernel(const float* __restrict__ O, const float* __restrict__ Y1,
                                const float* __restrict__ w, __bf16* __restrict__ Og) {
  __shared__ float red[256];
  int blk = blockIdx.x;            // bt*H + h
  int h = blk % HH, bt = blk / HH;
  int c = threadIdx.x;             // 0..255
  float o = O[(size_t)bt * VAL_DIM + h * VD + c];
  red[c] = o * o;
  for (int st = 128; st > 0; st >>= 1) {
    __syncthreads();
    if (c < st) red[c] += red[c + st];
  }
  __syncthreads();
  float rms = rsqrtf(red[0] / (float)VD + 1e-5f);
  float g = Y1[(size_t)bt * NCAT + 4096 + h * VD + c];
  float val = o * rms * w[c] * silu(g);
  Og[(size_t)bt * VAL_DIM + h * VD + c] = __builtin_bit_cast(__bf16, f2bf(val));
}

// ---------------- launch ----------------
extern "C" void kernel_launch(void* const* d_in, const int* in_sizes, int n_in,
                              void* d_out, int out_size, void* d_ws, size_t ws_size,
                              hipStream_t stream) {
  const float* x      = (const float*)d_in[0];
  const float* Wq     = (const float*)d_in[1];
  const float* Wk     = (const float*)d_in[2];
  const float* Wv     = (const float*)d_in[3];
  const float* Wa     = (const float*)d_in[4];
  const float* Wb     = (const float*)d_in[5];
  const float* Wg     = (const float*)d_in[6];
  const float* Wo     = (const float*)d_in[7];
  const float* A_log  = (const float*)d_in[8];
  const float* dtb    = (const float*)d_in[9];
  const float* convq  = (const float*)d_in[10];
  const float* convk  = (const float*)d_in[11];
  const float* convv  = (const float*)d_in[12];
  const float* onw    = (const float*)d_in[13];
  float* out = (float*)d_out;

  uint8_t* base = (uint8_t*)d_ws;
  size_t off = 0;
  auto alloc = [&](size_t bytes) -> void* {
    off = (off + 255) & ~(size_t)255;
    void* p = base + off;
    off += bytes;
    return p;
  };
  unsigned* WcatP = (unsigned*)alloc((size_t)(DD / 2) * NCAT * 4);      // packed panels
  unsigned* WoP   = (unsigned*)alloc((size_t)(VAL_DIM / 2) * DD * 4);
  __bf16*   xB    = (__bf16*)alloc((size_t)ROWS * DD * 2);
  __bf16*   OgB   = (__bf16*)alloc((size_t)ROWS * VAL_DIM * 2);
  float*    Y1    = (float*)alloc((size_t)ROWS * NCAT * 4);
  float*    qc    = (float*)alloc((size_t)ROWS * KEY_DIM * 4);
  float*    kc    = (float*)alloc((size_t)ROWS * KEY_DIM * 4);
  float*    vc    = (float*)alloc((size_t)ROWS * VAL_DIM * 4);
  float*    al    = (float*)alloc((size_t)ROWS * HH * 4);
  float*    be    = (float*)alloc((size_t)ROWS * HH * 4);
  float*    O     = (float*)alloc((size_t)ROWS * VAL_DIM * 4);
  (void)ws_size; (void)n_in; (void)in_sizes; (void)out_size;

  // weights -> packed bf16 K-pair panels (Wq|Wk|Wv|Wg fused into WcatP)
  pack_w<<<(DD/2*1024+255)/256, 256, 0, stream>>>(Wq, WcatP, DD, 1024, 0,    NCAT);
  pack_w<<<(DD/2*1024+255)/256, 256, 0, stream>>>(Wk, WcatP, DD, 1024, 1024, NCAT);
  pack_w<<<(DD/2*2048+255)/256, 256, 0, stream>>>(Wv, WcatP, DD, 2048, 2048, NCAT);
  pack_w<<<(DD/2*2048+255)/256, 256, 0, stream>>>(Wg, WcatP, DD, 2048, 4096, NCAT);
  pack_w<<<(VAL_DIM/2*DD+255)/256, 256, 0, stream>>>(Wo, WoP, VAL_DIM, DD, 0, DD);

  // x -> bf16
  cvt_bf16<<<(ROWS*DD+255)/256, 256, 0, stream>>>(x, xB, ROWS * DD);

  // fused projection GEMM: Y1[4096 x 6144] = x @ [Wq|Wk|Wv|Wg]
  gemm_bf16_wmma<<<dim3(NCAT/128, ROWS/128), 256, 0, stream>>>(xB, WcatP, Y1, ROWS, NCAT, DD);

  // gates
  ab_kernel<<<(ROWS*HH+255)/256, 256, 0, stream>>>(x, Wa, Wb, A_log, dtb, al, be);

  // conv + silu (+ l2 norm for q,k)
  conv_qk_kernel<<<ROWS*HH, 128, 0, stream>>>(Y1, 0,    convq, qc);
  conv_qk_kernel<<<ROWS*HH, 128, 0, stream>>>(Y1, 1024, convk, kc);
  conv_v_kernel<<<(ROWS*VAL_DIM+255)/256, 256, 0, stream>>>(Y1, convv, vc);

  // sequential delta-rule scan
  scan_kernel<<<BB*HH, 1024, 0, stream>>>(qc, kc, vc, al, be, O);

  // rms-norm + silu gate -> bf16
  gate_rms_kernel<<<ROWS*HH, 256, 0, stream>>>(O, Y1, onw, OgB);

  // output projection: out[4096 x 1024] = Og @ Wo
  gemm_bf16_wmma<<<dim3(DD/128, ROWS/128), 256, 0, stream>>>(OgB, WoP, out, ROWS, DD, VAL_DIM);
}
